// SelfAttentionBlock__14405320310906
// MI455X (gfx1250) — compile-verified, matrix operand order checked by reading, not run on hardware
//
#include <hip/hip_runtime.h>
#include <hip/hip_bf16.h>

// ---------------------------------------------------------------------------
// SelfAttentionBlock for MI455X (gfx1250, wave32, WMMA, async-to-LDS).
//   x[2,512,64,64] -> (context[2,256,4096], sim_map[2,4096,4096])
// GEMMs (Gram 17.2GF + context 17.2GF) on v_wmma_f32_16x16x32_f16.
// Block-shared A operands staged via GLOBAL_LOAD_ASYNC_TO_LDS_B128
// (ASYNCcnt) into the 320KB WGP LDS; per-wave B operands stream from L2
// with global_prefetch_b8 ahead. HBM floor is sim_map traffic (~0.5 GB,
// ~22us @ 23.3TB/s), so f16-in/f32-accumulate WMMA is roofline-free.
// ---------------------------------------------------------------------------

typedef __attribute__((ext_vector_type(16))) _Float16 v16h;
typedef __attribute__((ext_vector_type(8)))  _Float16 v8h;
typedef __attribute__((ext_vector_type(8)))  float    v8f;

#define BATCH 2
#define CIN   512
#define CK    256
#define CV    256
#define HH    64
#define WW    64
#define NN    4096   // H*W

// Per-lane A/B fragment for 16x16x32 f16 WMMA, NT layout (K contiguous).
// lane<16: K = {0..7, 16..23}; lane>=16: K = {8..15, 24..31}.
// `base` must already include the +8-half offset for high lanes.
static __device__ __forceinline__ v16h load_frag_h(const _Float16* __restrict__ base) {
    v16h r;
    v8h lo = *(const v8h*)(base);
    v8h hi = *(const v8h*)(base + 16);
#pragma unroll
    for (int i = 0; i < 8; ++i) { r[i] = lo[i]; r[i + 8] = hi[i]; }
    return r;
}

// 16B async copy global->LDS (ASYNCcnt-tracked, no VGPR staging).
static __device__ __forceinline__ void async_copy_b128(unsigned lds_byte_addr,
                                                       const void* gsrc) {
    unsigned long long ga = (unsigned long long)(size_t)gsrc;
    asm volatile("global_load_async_to_lds_b128 %0, %1, off"
                 :: "v"(lds_byte_addr), "v"(ga) : "memory");
}
static __device__ __forceinline__ void async_wait0() {
    asm volatile("s_wait_asynccnt 0x0" ::: "memory");
}

// ---------------------------------------------------------------------------
// K0: transpose wk [Ck,C] -> wkT [C,Ck] so feat kernel weight reads coalesce.
__global__ void wk_transpose_kernel(const float* __restrict__ wk,
                                    float* __restrict__ wkT) {
    int i = blockIdx.x * 256 + threadIdx.x;      // over Ck*C = 131072
    int c  = i / CIN;
    int cc = i % CIN;
    wkT[cc * CK + c] = wk[c * CIN + cc];
}

// ---------------------------------------------------------------------------
// K1: feat = ReLU(BN(wk @ x)), stored transposed f16: featT[b][n][ck].
__global__ void feat_kernel(const float* __restrict__ x,
                            const float* __restrict__ wkT,
                            const float* __restrict__ bk,
                            const float* __restrict__ gamma,
                            const float* __restrict__ beta,
                            const float* __restrict__ mean,
                            const float* __restrict__ var,
                            _Float16* __restrict__ featT) {
    int c = threadIdx.x;        // 0..255
    int n = blockIdx.x;         // 0..4095
    int b = blockIdx.y;
    const float* xp = x + (size_t)b * CIN * NN + n;
    float acc = 0.f;
#pragma unroll 4
    for (int cc = 0; cc < CIN; ++cc)
        acc += wkT[cc * CK + c] * xp[(size_t)cc * NN];
    float s = gamma[c] * rsqrtf(var[c] + 1e-5f);
    float o = (acc + bk[c] - mean[c]) * s + beta[c];
    o = fmaxf(o, 0.f);
    featT[((size_t)b * NN + n) * CK + c] = (_Float16)o;
}

// ---------------------------------------------------------------------------
// K2: Gram S[m,n] = scale * <featT[m,:], featT[n,:]>  via WMMA f16.
// 8 waves/block share one A tile (rows m0..m0+15, 8KB) staged async->LDS.
__global__ void gram_kernel(const _Float16* __restrict__ featT,
                            float* __restrict__ sim) {
    __shared__ __align__(16) _Float16 atile[16 * CK];   // 8 KB
    int t    = threadIdx.x;
    int lane = t & 31;
    int wave = t >> 5;
    int b  = blockIdx.z;
    int m0 = blockIdx.y * 16;
    int n0 = (blockIdx.x * 8 + wave) * 16;
    const _Float16* base = featT + (size_t)b * NN * CK;

    // async-stage A tile: 16 rows x 512B = 512 x 16B chunks, 2 per thread
    {
        const char* gsrc = (const char*)(base + (size_t)m0 * CK);
        unsigned lds0 = (unsigned)(size_t)(&atile[0]);
#pragma unroll
        for (int i = 0; i < 2; ++i) {
            int chunk = t + i * 256;            // 0..511
            int row   = chunk >> 5;             // 32 chunks (512B) per row
            int off   = (chunk & 31) * 16;      // bytes within row
            async_copy_b128(lds0 + (unsigned)(row * 512 + off),
                            gsrc + (size_t)row * (CK * 2) + off);
        }
        async_wait0();
        __syncthreads();
    }

    int sel = (lane >> 4) * 8;
    const _Float16* alds = atile + (size_t)(lane & 15) * CK + sel;
    const _Float16* bp   = base + (size_t)(n0 + (lane & 15)) * CK + sel;
    v8f acc = {};
#pragma unroll
    for (int k = 0; k < CK; k += 32) {
        __builtin_prefetch(bp + k + 32, 0, 1);           // global_prefetch_b8
        v16h a  = load_frag_h(alds + k);                 // ds_load_b128 x2
        v16h bb = load_frag_h(bp + k);                   // global_load_b128 x2
        acc = __builtin_amdgcn_wmma_f32_16x16x32_f16(
            false, a, false, bb, (short)0, acc, false, false);
    }
    const float scale = 0.0625f;                         // 256^-0.5
    float* out = sim + (size_t)b * NN * NN;
    int col = n0 + (lane & 15);
    int row = m0 + (lane >> 4) * 8;
#pragma unroll
    for (int r = 0; r < 8; ++r)
        out[(size_t)(row + r) * NN + col] = acc[r] * scale;
}

// ---------------------------------------------------------------------------
// K3: row softmax in place. One block per (b,row); 256 threads x 16 elems.
__global__ void softmax_kernel(float* __restrict__ sim) {
    int t = threadIdx.x;
    float* p = sim + ((size_t)blockIdx.y * NN + blockIdx.x) * NN;
    float vals[16];
    float m = -3.4e38f;
#pragma unroll
    for (int i = 0; i < 16; ++i) { vals[i] = p[t + i * 256]; m = fmaxf(m, vals[i]); }
    __shared__ float red[256];
    red[t] = m; __syncthreads();
    for (int s = 128; s > 0; s >>= 1) {
        if (t < s) red[t] = fmaxf(red[t], red[t + s]);
        __syncthreads();
    }
    m = red[0]; __syncthreads();
    float sum = 0.f;
#pragma unroll
    for (int i = 0; i < 16; ++i) { vals[i] = __expf(vals[i] - m); sum += vals[i]; }
    red[t] = sum; __syncthreads();
    for (int s = 128; s > 0; s >>= 1) {
        if (t < s) red[t] += red[t + s];
        __syncthreads();
    }
    float inv = 1.f / red[0];
#pragma unroll
    for (int i = 0; i < 16; ++i) p[t + i * 256] = vals[i] * inv;
}

// ---------------------------------------------------------------------------
// K4: direct conv3x3 + ReLU -> v1 f32 [b][co][n]; x is L2-resident.
__global__ void conv3_kernel(const float* __restrict__ x,
                             const float* __restrict__ wv1,
                             const float* __restrict__ bv1,
                             float* __restrict__ v1) {
    int w  = threadIdx.x;
    int h  = blockIdx.x * blockDim.y + threadIdx.y;
    int co = blockIdx.y;
    int b  = blockIdx.z;
    const float* xb = x + (size_t)b * CIN * NN;
    const float* wp = wv1 + (size_t)co * CIN * 9;
    float acc = bv1[co];
    for (int ci = 0; ci < CIN; ++ci) {
        const float* xc = xb + (size_t)ci * NN;
        const float* wc = wp + ci * 9;
#pragma unroll
        for (int dh = -1; dh <= 1; ++dh) {
            int hh = h + dh;
            if ((unsigned)hh >= (unsigned)HH) continue;
#pragma unroll
            for (int dw = -1; dw <= 1; ++dw) {
                int ww = w + dw;
                if ((unsigned)ww >= (unsigned)WW) continue;
                acc += wc[(dh + 1) * 3 + (dw + 1)] * xc[hh * WW + ww];
            }
        }
    }
    v1[((size_t)b * CV + co) * NN + h * WW + w] = fmaxf(acc, 0.f);
}

// ---------------------------------------------------------------------------
// K5: value 1x1 conv + ReLU -> f16 v[b][c][n] (n contiguous: WMMA A rows).
__global__ void value1x1_kernel(const float* __restrict__ v1,
                                const float* __restrict__ wv2,
                                const float* __restrict__ bv2,
                                _Float16* __restrict__ vh) {
    int n = blockIdx.x * 256 + threadIdx.x;
    int c = blockIdx.y;
    int b = blockIdx.z;
    const float* vp = v1 + (size_t)b * CV * NN + n;
    const float* wp = wv2 + c * CV;               // uniform -> scalar loads
    float acc = bv2[c];
#pragma unroll 4
    for (int cc = 0; cc < CV; ++cc)
        acc += wp[cc] * vp[(size_t)cc * NN];
    vh[((size_t)b * CV + c) * NN + n] = (_Float16)fmaxf(acc, 0.f);
}

// ---------------------------------------------------------------------------
// K6: context[c,n] = sum_m v[c,m] * sim[n,m]  via WMMA f16.
// Shared A panel (16 v-rows) staged async->LDS in 32KB chunks of K=1024;
// B rows stream from sim f32 and convert with v_cvt_pk_rtz_f16_f32.
#define KPANEL 1024
__global__ void context_kernel(const _Float16* __restrict__ vh,
                               const float* __restrict__ sim,
                               float* __restrict__ ctx) {
    __shared__ __align__(16) _Float16 atile[16 * KPANEL];   // 32 KB
    int t    = threadIdx.x;
    int lane = t & 31;
    int wave = t >> 5;
    int b  = blockIdx.z;
    int c0 = blockIdx.y * 16;
    int n0 = (blockIdx.x * 8 + wave) * 16;
    int sel = (lane >> 4) * 8;
    const _Float16* arow = vh + ((size_t)b * CV + c0) * NN;  // 16-row panel base
    const float*    bp   = sim + ((size_t)b * NN + n0 + (lane & 15)) * NN + sel;
    const _Float16* alds = atile + (size_t)(lane & 15) * KPANEL + sel;
    unsigned lds0 = (unsigned)(size_t)(&atile[0]);
    v8f acc = {};

    for (int kp = 0; kp < NN; kp += KPANEL) {
        // async-stage 16 rows x 2KB = 2048 x 16B chunks, 8 per thread
#pragma unroll
        for (int i = 0; i < 8; ++i) {
            int chunk = t + i * 256;            // 0..2047
            int row   = chunk >> 7;             // 128 chunks (2KB) per row
            int off   = (chunk & 127) * 16;     // bytes within row
            async_copy_b128(lds0 + (unsigned)(row * (KPANEL * 2) + off),
                            (const char*)(arow + (size_t)row * NN + kp) + off);
        }
        async_wait0();
        __syncthreads();

#pragma unroll 4
        for (int k = 0; k < KPANEL; k += 32) {
            __builtin_prefetch(bp + kp + k + 32, 0, 1);
            v16h a = load_frag_h(alds + k);      // ds_load_b128 x2
            v16h bb;
            const float* bq = bp + kp + k;
#pragma unroll
            for (int i = 0; i < 4; ++i) {        // v_cvt_pk_rtz_f16_f32
                auto p = __builtin_amdgcn_cvt_pkrtz(bq[2 * i], bq[2 * i + 1]);
                bb[2 * i]     = (_Float16)p[0];
                bb[2 * i + 1] = (_Float16)p[1];
            }
#pragma unroll
            for (int i = 0; i < 4; ++i) {
                auto p = __builtin_amdgcn_cvt_pkrtz(bq[16 + 2 * i], bq[17 + 2 * i]);
                bb[8 + 2 * i] = (_Float16)p[0];
                bb[9 + 2 * i] = (_Float16)p[1];
            }
            acc = __builtin_amdgcn_wmma_f32_16x16x32_f16(
                false, a, false, bb, (short)0, acc, false, false);
        }
        __syncthreads();                         // before next panel overwrite
    }

    float* out = ctx + (size_t)b * CV * NN;
    int nn = n0 + (lane & 15);
    int cb = c0 + (lane >> 4) * 8;
#pragma unroll
    for (int r = 0; r < 8; ++r)
        out[(size_t)(cb + r) * NN + nn] = acc[r];
}

// ---------------------------------------------------------------------------
extern "C" void kernel_launch(void* const* d_in, const int* in_sizes, int n_in,
                              void* d_out, int out_size, void* d_ws, size_t ws_size,
                              hipStream_t stream) {
    (void)in_sizes; (void)n_in; (void)out_size; (void)ws_size;
    const float* x     = (const float*)d_in[0];
    const float* wk    = (const float*)d_in[1];
    const float* bk    = (const float*)d_in[2];
    const float* gamma = (const float*)d_in[3];
    const float* beta  = (const float*)d_in[4];
    const float* mean  = (const float*)d_in[5];
    const float* var   = (const float*)d_in[6];
    const float* wv1   = (const float*)d_in[7];
    const float* bv1   = (const float*)d_in[8];
    const float* wv2   = (const float*)d_in[9];
    const float* bv2   = (const float*)d_in[10];

    float* ctx = (float*)d_out;                       // [2,256,4096]
    float* sim = ctx + (size_t)BATCH * CV * NN;       // [2,4096,4096]

    // workspace carve-up (16-byte aligned blocks)
    char* ws = (char*)d_ws;
    float*    wkT   = (float*)(ws);                                  // 0.5 MB
    _Float16* featT = (_Float16*)(ws + (size_t)524288);              // 4 MB
    float*    v1    = (float*)(ws + (size_t)524288 + 4194304);       // 8 MB
    _Float16* vh    = (_Float16*)(ws + (size_t)524288 + 4194304 + 8388608); // 4 MB

    // key/query path
    wk_transpose_kernel<<<dim3((CK * CIN) / 256), 256, 0, stream>>>(wk, wkT);
    feat_kernel<<<dim3(NN, BATCH), 256, 0, stream>>>(x, wkT, bk, gamma, beta,
                                                     mean, var, featT);
    // attention logits + softmax (WMMA + async LDS)
    gram_kernel<<<dim3(NN / 128, NN / 16, BATCH), 256, 0, stream>>>(featT, sim);
    softmax_kernel<<<dim3(NN, BATCH), 256, 0, stream>>>(sim);
    // value path
    conv3_kernel<<<dim3(HH / 4, CV, BATCH), dim3(WW, 4), 0, stream>>>(x, wv1, bv1, v1);
    value1x1_kernel<<<dim3(NN / 256, CV, BATCH), 256, 0, stream>>>(v1, wv2, bv2, vh);
    // context aggregation (WMMA + async LDS panels)
    context_kernel<<<dim3(NN / 128, CV / 16, BATCH), 256, 0, stream>>>(vh, sim, ctx);
}
